// MtanRNNRegr_56143812493711
// MI455X (gfx1250) — matrix operand order repaired and checked by reading, not compiled, not run on hardware
//
#include <hip/hip_runtime.h>
#include <hip/hip_bf16.h>
#include <math.h>

// MtanRNNRegr on gfx1250: WMMA f32<=f16 for all GEMMs, factored masked softmax,
// single-workgroup fp32 recurrence. Latency-bound problem (~0.7 GFLOP, ~7MB).
// Round 2: all WMMA operands vectorized -- A frags = 2x b128 (row-major is
// already lane-contiguous), B frags = 2x b128 via fragment-major pre-swizzle,
// E written as b128 stores. Adds global_prefetch on B panels.

typedef __attribute__((ext_vector_type(16))) _Float16 v16h;
typedef __attribute__((ext_vector_type(8)))  _Float16 v8h;
typedef __attribute__((ext_vector_type(8)))  float    v8f;

#define B_   16
#define L_   256
#define LQ_  64
#define DIN_ 32
#define ET_  128
#define H_   4
#define ETK_ 32
#define DIM_ 64

// ---------- WMMA fragment helpers (16x16x32 f16, k = 16*(lane/16) + s in both A and B) ----------

__device__ __forceinline__ v8f wmma_f16(v16h a, v16h b, v8f c) {
    return __builtin_amdgcn_wmma_f32_16x16x32_f16(
        /*neg_a=*/false, a, /*neg_b=*/false, b,
        /*c_mod=*/(short)0, c, /*reuse_a=*/false, /*reuse_b=*/false);
}

// A: 16x32 f16 tile, row-major with leading dim lda (elements, multiple of 8).
// Lane l<16 (M=l): halves 0..15 -> K=0..15.
// Lane l>=16:      halves 0..15 -> K=16..31.
// Both runs are contiguous in memory -> two global_load_b128 per lane.
__device__ __forceinline__ v16h load_a_frag(const _Float16* A, int lda) {
    int lane = threadIdx.x & 31;
    int m = lane & 15, hi = lane >> 4;
    const _Float16* p = A + m * lda + hi * 16;
    v8h lo  = *(const v8h*)(p);
    v8h hh  = *(const v8h*)(p + 8);
    return __builtin_shufflevector(lo, hh, 0, 1, 2, 3, 4, 5, 6, 7,
                                            8, 9, 10, 11, 12, 13, 14, 15);
}

// B: 32x16 tile pre-swizzled fragment-major: tile = 512 halves,
// lane's 16 halves contiguous at tile + lane*16.
// Element (k, n): lane = (k>>4)*16 + n, half = k & 15.
__device__ __forceinline__ v16h load_b_frag_sw(const _Float16* tile) {
    int lane = threadIdx.x & 31;
    const _Float16* p = tile + lane * 16;
    v8h lo = *(const v8h*)(p);
    v8h hh = *(const v8h*)(p + 8);
    return __builtin_shufflevector(lo, hh, 0, 1, 2, 3, 4, 5, 6, 7,
                                            8, 9, 10, 11, 12, 13, 14, 15);
}

// ---------- K0: f16-swizzle weights, time embeddings, masked value matrices ----------

__device__ __forceinline__ void swizzle_weight_f16(const float* W, _Float16* out,
                                                   int K, int N, int tid, int nthr) {
    // out[((kt*(N/16)+nt)*32 + lane)*16 + h] = W[(kt*32 + (lane>>4)*16 + h)*N + nt*16 + (lane&15)]
    int total = K * N, ntn = N >> 4;
    for (int s = tid; s < total; s += nthr) {
        int h = s & 15, lane = (s >> 4) & 31, tile = s >> 9;
        int ntile = tile % ntn, ktile = tile / ntn;
        int k = ktile * 32 + (lane >> 4) * 16 + h;
        int col = ntile * 16 + (lane & 15);
        out[s] = (_Float16)W[k * N + col];
    }
}

__global__ void prep_kernel(const float* x, const float* ts, const float* mask, const float* query,
                            const float* w_lin, const float* b_lin, const float* w_per, const float* b_per,
                            const float* Wq, const float* Wk, const float* Wo, const float* Wih,
                            _Float16* wq_s, _Float16* wk_s, _Float16* wo_s, _Float16* wih_s,
                            _Float16* kemb_h, _Float16* qemb_h, _Float16* V1, _Float16* Vm) {
    int tid = blockIdx.x * blockDim.x + threadIdx.x;
    int nt  = gridDim.x * blockDim.x;
    swizzle_weight_f16(Wq,  wq_s,  ET_,     ET_, tid, nt);
    swizzle_weight_f16(Wk,  wk_s,  ET_,     ET_, tid, nt);
    swizzle_weight_f16(Wih, wih_s, ET_,     ET_, tid, nt);
    swizzle_weight_f16(Wo,  wo_s,  2 * ET_, ET_, tid, nt);
    // kemb: (B*L, ET) row-major f16 (A operand); col 0 linear, cols 1.. = sin
    for (int i = tid; i < B_ * L_ * ET_; i += nt) {
        int row = i >> 7, c = i & 127;
        float t = ts[row];
        float v = (c == 0) ? fmaf(t, w_lin[0], b_lin[0])
                           : sinf(fmaf(t, w_per[c - 1], b_per[c - 1]));
        kemb_h[i] = (_Float16)v;
    }
    for (int i = tid; i < LQ_ * ET_; i += nt) {
        int row = i >> 7, c = i & 127;
        float t = query[row];
        float v = (c == 0) ? fmaf(t, w_lin[0], b_lin[0])
                           : sinf(fmaf(t, w_per[c - 1], b_per[c - 1]));
        qemb_h[i] = (_Float16)v;
    }
    // V1 = m2*xc (second half: mask*mask = mask), Vm = m2.
    // Swizzled per batch as B operand: per b, K=L (rows l), N=DIM (cols d);
    // 16384 halves per b; tile grid (kt=0..7, nt=0..3).
    for (int j = tid; j < B_ * L_ * DIM_; j += nt) {
        int b = j >> 14, s = j & 16383;
        int h = s & 15, lane = (s >> 4) & 31, tile = s >> 9;
        int ntile = tile & 3, ktile = tile >> 2;
        int l = ktile * 32 + (lane >> 4) * 16 + h;
        int d = ntile * 16 + (lane & 15);
        float m = mask[(b * L_ + l) * DIN_ + (d & 31)];
        float v = (d < DIN_) ? m * x[(b * L_ + l) * DIN_ + d] : m;
        V1[j] = (_Float16)v;
        Vm[j] = (_Float16)m;
    }
}

// ---------- K1: k = kemb @ Wk + bk, scattered to swizzled kT (B operand of scores) ----------
// kT per (b,h): K=ETK(32) x N=L(256) -> 16 tiles of 512 halves; block base bh*8192.

__global__ void proj_k_kernel(const _Float16* kemb, const _Float16* wk_s, const float* bk,
                              _Float16* kT) {
    int wid = blockIdx.x * 8 + (threadIdx.x >> 5);   // 2048 waves: 256 row-tiles x 8 col-tiles
    int rt = wid >> 3, ct = wid & 7;
    const _Float16* Abase = kemb + rt * 16 * ET_;
    const _Float16* Btile = wk_s + ct * 512;          // + kk*8*512 per K step
    __builtin_prefetch(Btile, 0, 3);
    v8f acc = {};
#pragma unroll
    for (int kk = 0; kk < 4; ++kk) {
        v16h a = load_a_frag(Abase + kk * 32, ET_);
        v16h b = load_b_frag_sw(Btile + kk * 8 * 512);
        acc = wmma_f16(a, b, acc);
    }
    int lane = threadIdx.x & 31, n = lane & 15, hi = lane >> 4;
    int col = ct * 16 + n;                            // ET column
    int head = col >> 5, e = col & 31;                // k-dim index of kT
    int hb = e & 15, hib = e >> 4;
    float bias = bk[col];
#pragma unroll
    for (int r = 0; r < 8; ++r) {
        int row = rt * 16 + r + hi * 8;               // row in [0, B*L)
        int bidx = row >> 8, l = row & 255;
        int bh = bidx * H_ + head;
        kT[bh * 8192 + (((l >> 4) * 32 + hib * 16 + (l & 15)) << 4) + hb] =
            (_Float16)(acc[r] + bias);
    }
}

// ---------- K2: q = qemb @ Wq + bq -> (h, q, e) row-major (A operand of scores) ----------

__global__ void proj_q_kernel(const _Float16* qemb, const _Float16* wq_s, const float* bq,
                              _Float16* q_h) {
    int wid = blockIdx.x * 8 + (threadIdx.x >> 5);   // 32 waves: 4 row-tiles x 8 col-tiles
    int rt = wid >> 3, ct = wid & 7;
    const _Float16* Abase = qemb + rt * 16 * ET_;
    const _Float16* Btile = wq_s + ct * 512;
    v8f acc = {};
#pragma unroll
    for (int kk = 0; kk < 4; ++kk) {
        v16h a = load_a_frag(Abase + kk * 32, ET_);
        v16h b = load_b_frag_sw(Btile + kk * 8 * 512);
        acc = wmma_f16(a, b, acc);
    }
    int lane = threadIdx.x & 31, n = lane & 15, hi = lane >> 4;
    int col = ct * 16 + n;
    int head = col >> 5, e = col & 31;
    float bias = bq[col];
#pragma unroll
    for (int r = 0; r < 8; ++r) {
        int row = rt * 16 + r + hi * 8;               // q index
        q_h[(head * LQ_ + row) * ETK_ + e] = (_Float16)(acc[r] + bias);
    }
}

// ---------- K3: scores (WMMA, K=32) + row max + E = exp(s - max), b128 stores ----------
// grid = B*H*4 blocks (one 16-row tile of one (b,h)); block = 512 threads = 16 waves.

__global__ void scores_kernel(const _Float16* q_h, const _Float16* kT, _Float16* E) {
    __shared__ float S[16][L_];
    int bh = blockIdx.x >> 2;                         // b*H + h
    int rt = blockIdx.x & 3;
    int h  = bh & 3;
    int w    = threadIdx.x >> 5;                      // 0..15 = column tile = nt
    int lane = threadIdx.x & 31;
    const _Float16* Abase = q_h + (h * LQ_ + rt * 16) * ETK_;
    const _Float16* Btile = kT + bh * 8192 + w * 512;
    v16h a = load_a_frag(Abase, ETK_);
    v16h b = load_b_frag_sw(Btile);
    v8f acc = {};
    acc = wmma_f16(a, b, acc);
    int n = lane & 15, hi = lane >> 4;
    const float scale = 0.17677669529663687f;         // 1/sqrt(ETK)
#pragma unroll
    for (int r = 0; r < 8; ++r)
        S[r + hi * 8][w * 16 + n] = acc[r] * scale;
    __syncthreads();
    // wave w reduces row w; each lane owns 8 contiguous columns -> one b128 store
    int c0 = lane * 8;
    float m = -1e30f;
#pragma unroll
    for (int i = 0; i < 8; ++i) m = fmaxf(m, S[w][c0 + i]);
#pragma unroll
    for (int off = 16; off > 0; off >>= 1) m = fmaxf(m, __shfl_xor(m, off, 32));
    int qrow = rt * 16 + w;
    v8h ev;
#pragma unroll
    for (int i = 0; i < 8; ++i) ev[i] = (_Float16)__expf(S[w][c0 + i] - m);
    *(v8h*)(E + (bh * LQ_ + qrow) * L_ + c0) = ev;
}

// ---------- K4: num = E @ V1, den = E @ Vm (K=256), att = num/den ----------
// grid = B*H blocks; block = 512 threads = 16 waves = 4x4 output tiles of 64x64.

__global__ void attn_kernel(const _Float16* E, const _Float16* V1, const _Float16* Vm,
                            _Float16* att) {
    int bh = blockIdx.x;
    int b = bh >> 2, h = bh & 3;
    int w = threadIdx.x >> 5;
    int rt = w >> 2, ct = w & 3;
    const _Float16* Abase = E + (bh * LQ_ + rt * 16) * L_;
    const _Float16* V1b   = V1 + b * 16384 + ct * 512;   // + kk*4*512 per K step
    const _Float16* Vmb   = Vm + b * 16384 + ct * 512;
    __builtin_prefetch(V1b, 0, 3);
    __builtin_prefetch(Vmb, 0, 3);
    v8f accN = {}, accD = {};
#pragma unroll
    for (int kk = 0; kk < 8; ++kk) {
        v16h a  = load_a_frag(Abase + kk * 32, L_);
        v16h b1 = load_b_frag_sw(V1b + kk * 4 * 512);
        accN = wmma_f16(a, b1, accN);
        v16h b2 = load_b_frag_sw(Vmb + kk * 4 * 512);
        accD = wmma_f16(a, b2, accD);
    }
    int lane = threadIdx.x & 31, n = lane & 15, hi = lane >> 4;
#pragma unroll
    for (int r = 0; r < 8; ++r) {
        int q = rt * 16 + r + hi * 8;
        int d = ct * 16 + n;
        float den = accD[r];
        float val = (den > 0.f) ? accN[r] / den : 0.f;   // all-masked column: measure-zero
        att[(b * LQ_ + q) * (H_ * DIM_) + h * DIM_ + d] = (_Float16)val;
    }
}

// ---------- K5: out = att(1024x256) @ Wo + bo -> f16 ----------

__global__ void outproj_kernel(const _Float16* att, const _Float16* wo_s, const float* bo,
                               _Float16* outh) {
    int wid = blockIdx.x * 8 + (threadIdx.x >> 5);    // 512 waves: 64 row-tiles x 8 col-tiles
    int rt = wid >> 3, ct = wid & 7;
    const _Float16* Abase = att + rt * 16 * (H_ * DIM_);
    const _Float16* Btile = wo_s + ct * 512;
    __builtin_prefetch(Btile, 0, 3);
    v8f acc = {};
#pragma unroll
    for (int kk = 0; kk < 8; ++kk) {
        v16h a = load_a_frag(Abase + kk * 32, H_ * DIM_);
        v16h b = load_b_frag_sw(Btile + kk * 8 * 512);
        acc = wmma_f16(a, b, acc);
    }
    int lane = threadIdx.x & 31, n = lane & 15, hi = lane >> 4;
    int col = ct * 16 + n;
    float bias = bo[col];
#pragma unroll
    for (int r = 0; r < 8; ++r) {
        int row = rt * 16 + r + hi * 8;
        outh[row * ET_ + col] = (_Float16)(acc[r] + bias);
    }
}

// ---------- K6: G[t][b][:] = out[b,t,:] @ W_ih + b_ih (t-independent precompute) ----------

__global__ void rnnin_kernel(const _Float16* outh, const _Float16* wih_s, const float* b_ih,
                             float* G) {
    int wid = blockIdx.x * 8 + (threadIdx.x >> 5);    // 512 waves
    int rt = wid >> 3, ct = wid & 7;
    const _Float16* Abase = outh + rt * 16 * ET_;
    const _Float16* Btile = wih_s + ct * 512;
    v8f acc = {};
#pragma unroll
    for (int kk = 0; kk < 4; ++kk) {
        v16h a = load_a_frag(Abase + kk * 32, ET_);
        v16h b = load_b_frag_sw(Btile + kk * 8 * 512);
        acc = wmma_f16(a, b, acc);
    }
    int lane = threadIdx.x & 31, n = lane & 15, hi = lane >> 4;
    int col = ct * 16 + n;
    float bias = b_ih[col];
#pragma unroll
    for (int r = 0; r < 8; ++r) {
        int row = rt * 16 + r + hi * 8;               // row = b*LQ + t
        int b = row >> 6, t = row & 63;
        G[(t * B_ + b) * ET_ + col] = acc[r] + bias;
    }
}

// ---------- K7: sequential tanh-RNN over LQ steps + 4-layer regressor (fp32, 1 WG) ----------

__global__ void rnn_reg_kernel(const float* G, const float* W_hh, const float* b_hh,
                               const float* r1w, const float* r1b,
                               const float* r2w, const float* r2b,
                               const float* r3w, const float* r3b,
                               const float* r4w, const float* r4b,
                               float* out) {
    __shared__ float hbuf[2][B_][ET_];
    int tid = threadIdx.x;                            // blockDim == 256
    for (int i = tid; i < B_ * ET_; i += 256) hbuf[0][i >> 7][i & 127] = 0.f;
    __syncthreads();
    int cur = 0;
    for (int t = 0; t < LQ_; ++t) {
        for (int o = tid; o < B_ * ET_; o += 256) {
            int b = o >> 7, c = o & 127;
            float acc = G[(t * B_ + b) * ET_ + c] + b_hh[c];
            const float* hrow = hbuf[cur][b];
#pragma unroll 4
            for (int e = 0; e < ET_; ++e) acc = fmaf(hrow[e], W_hh[e * ET_ + c], acc);
            hbuf[1 - cur][b][c] = tanhf(acc);
        }
        __syncthreads();
        cur ^= 1;
    }
    const float* Ws[3] = {r1w, r2w, r3w};
    const float* Bs[3] = {r1b, r2b, r3b};
    for (int layer = 0; layer < 3; ++layer) {
        const float* W = Ws[layer];
        const float* bias = Bs[layer];
        for (int o = tid; o < B_ * ET_; o += 256) {
            int b = o >> 7, c = o & 127;
            float acc = bias[c];
            const float* hrow = hbuf[cur][b];
#pragma unroll 4
            for (int e = 0; e < ET_; ++e) acc = fmaf(hrow[e], W[e * ET_ + c], acc);
            hbuf[1 - cur][b][c] = acc;
        }
        __syncthreads();
        cur ^= 1;
    }
    if (tid < B_ * 8) {
        int b = tid >> 3, j = tid & 7;
        float acc = r4b[j];
        const float* hrow = hbuf[cur][b];
#pragma unroll 4
        for (int e = 0; e < ET_; ++e) acc = fmaf(hrow[e], r4w[e * 8 + j], acc);
        out[b * 8 + j] = acc;
    }
}

// ---------- host ----------

extern "C" void kernel_launch(void* const* d_in, const int* in_sizes, int n_in,
                              void* d_out, int out_size, void* d_ws, size_t ws_size,
                              hipStream_t stream) {
    const float* x    = (const float*)d_in[0];
    const float* ts   = (const float*)d_in[1];
    const float* mask = (const float*)d_in[2];
    const float* qry  = (const float*)d_in[3];
    const float* wlin = (const float*)d_in[4];
    const float* blin = (const float*)d_in[5];
    const float* wper = (const float*)d_in[6];
    const float* bper = (const float*)d_in[7];
    const float* Wq   = (const float*)d_in[8];
    const float* bq   = (const float*)d_in[9];
    const float* Wk   = (const float*)d_in[10];
    const float* bk   = (const float*)d_in[11];
    const float* Wo   = (const float*)d_in[12];
    const float* bo   = (const float*)d_in[13];
    const float* Wih  = (const float*)d_in[14];
    const float* bih  = (const float*)d_in[15];
    const float* Whh  = (const float*)d_in[16];
    const float* bhh  = (const float*)d_in[17];
    const float* r1w  = (const float*)d_in[18];
    const float* r1b  = (const float*)d_in[19];
    const float* r2w  = (const float*)d_in[20];
    const float* r2b  = (const float*)d_in[21];
    const float* r3w  = (const float*)d_in[22];
    const float* r3b  = (const float*)d_in[23];
    const float* r4w  = (const float*)d_in[24];
    const float* r4b  = (const float*)d_in[25];

    char* ws = (char*)d_ws;
    size_t off = 0;
    auto carve = [&](size_t bytes) -> char* {
        char* p = ws + off;
        off += (bytes + 255) & ~(size_t)255;
        return p;
    };
    _Float16* wq_s   = (_Float16*)carve(ET_ * ET_ * 2);
    _Float16* wk_s   = (_Float16*)carve(ET_ * ET_ * 2);
    _Float16* wih_s  = (_Float16*)carve(ET_ * ET_ * 2);
    _Float16* wo_s   = (_Float16*)carve(2 * ET_ * ET_ * 2);
    _Float16* kemb_h = (_Float16*)carve((size_t)B_ * L_ * ET_ * 2);
    _Float16* qemb_h = (_Float16*)carve(LQ_ * ET_ * 2);
    _Float16* kT_h   = (_Float16*)carve((size_t)B_ * H_ * ETK_ * L_ * 2);
    _Float16* q_h    = (_Float16*)carve(H_ * LQ_ * ETK_ * 2);
    _Float16* E_h    = (_Float16*)carve((size_t)B_ * H_ * LQ_ * L_ * 2);
    _Float16* V1_h   = (_Float16*)carve((size_t)B_ * L_ * DIM_ * 2);
    _Float16* Vm_h   = (_Float16*)carve((size_t)B_ * L_ * DIM_ * 2);
    _Float16* att_h  = (_Float16*)carve((size_t)B_ * LQ_ * H_ * DIM_ * 2);
    _Float16* out_h  = (_Float16*)carve((size_t)B_ * LQ_ * ET_ * 2);
    float*    G_f    = (float*)carve((size_t)LQ_ * B_ * ET_ * 4);

    prep_kernel<<<512, 256, 0, stream>>>(x, ts, mask, qry, wlin, blin, wper, bper,
                                         Wq, Wk, Wo, Wih,
                                         wq_s, wk_s, wo_s, wih_s,
                                         kemb_h, qemb_h, V1_h, Vm_h);
    proj_k_kernel<<<256, 256, 0, stream>>>(kemb_h, wk_s, bk, kT_h);
    proj_q_kernel<<<4, 256, 0, stream>>>(qemb_h, wq_s, bq, q_h);
    scores_kernel<<<B_ * H_ * 4, 512, 0, stream>>>(q_h, kT_h, E_h);
    attn_kernel<<<B_ * H_, 512, 0, stream>>>(E_h, V1_h, Vm_h, att_h);
    outproj_kernel<<<64, 256, 0, stream>>>(att_h, wo_s, bo, out_h);
    rnnin_kernel<<<64, 256, 0, stream>>>(out_h, wih_s, bih, G_f);
    rnn_reg_kernel<<<1, 256, 0, stream>>>(G_f, Whh, bhh,
                                          r1w, r1b, r2w, r2b, r3w, r3b, r4w, r4b,
                                          (float*)d_out);
}
